// AirObject_47236050322062
// MI455X (gfx1250) — compile-verified, matrix operand order checked by reading, not run on hardware
//
#include <hip/hip_runtime.h>
#include <hip/hip_bf16.h>

// ---------------------------------------------------------------------------
// CDNA5 / gfx1250 implementation of the AirObject forward pass.
// All dense GEMMs run on v_wmma_f32_16x16x32_f16 (wave32, fp32 accumulate),
// 128x64 macro tiles, 32x32 per wave (4 WMMAs per K-step), double-buffered
// LDS staging (b128 global loads / b128 LDS stores) with global_prefetch
// pipelining. The TCN conv+mean is algebraically collapsed to a column-sum
// + GEMV (mean commutes with convolution): 103 GFLOP -> ~13 MFLOP.
// ---------------------------------------------------------------------------

typedef __attribute__((ext_vector_type(16))) _Float16 v16h;
typedef __attribute__((ext_vector_type(8)))  float    v8f;

union Frag16 { v16h v; uint32_t u[8]; _Float16 h[16]; };
union Q16 { uint4 q; _Float16 h[8]; };

#define EPS_BN 1e-5f

// ---------------------------------------------------------------------------
// Generic WMMA GEMM:  C(f16)[M,N] = act( A(f16)[M,K] @ B[K,N] + bias )
//   - A always f16 row-major (lda)
//   - B either f32 (Bf32) or f16 (Bf16) row-major (ldb); exactly one non-null
//   - batched via blockIdx.z with element strides sA/sB/sC
//   - act: 0=none 1=relu 2=elu 3=relu-then-multiply-into-existing-C
// Macro tile 128x64, 256 threads = 8 waves in a 4x2 grid, each wave 32x32.
// BK=32 (one WMMA K-depth). Double-buffered LDS, one barrier per K-step.
// All M multiples of 128, N multiples of 64, K multiples of 32 here;
// lda/ldb multiples of 8 halfs -> every staged access is 16B aligned.
// ---------------------------------------------------------------------------
__global__ __launch_bounds__(256)
void k_gemm(const _Float16* __restrict__ A,
            const float*    __restrict__ Bf32,
            const _Float16* __restrict__ Bf16,
            const float*    __restrict__ bias,
            _Float16*       __restrict__ C,
            int M, int N, int K, int lda, int ldb, int ldc,
            long long sA, long long sB, long long sC, int act) {
  __shared__ __align__(16) _Float16 AsAll[2][128 * 32];  // [buf][row][k]
  __shared__ __align__(16) _Float16 BsAll[2][64 * 32];   // [buf][n][k] transposed

  const int z = blockIdx.z;
  A += (long long)z * sA;
  if (Bf32) Bf32 += (long long)z * sB; else Bf16 += (long long)z * sB;
  C += (long long)z * sC;

  const int tid   = threadIdx.x;
  const int tileM = blockIdx.y * 128;
  const int tileN = blockIdx.x * 64;

  const int lane = tid & 31;
  const int wid  = tid >> 5;
  const int rm0  = (wid >> 1) * 32;       // wave row base (0/32/64/96)
  const int cn0  = (wid & 1) * 32;        // wave col base (0/32)
  const int lrow = lane & 15;
  const int kb2  = (lane < 16) ? 0 : 4;   // u32 offset: K base 0 or 8 halves

  // ---- staging decomposition: contiguous 128-bit chunks per thread ----
  // A: 128 rows x 32 halfs; thread owns row=tid>>1, halfs [(tid&1)*16, +16).
  const int aRow = tid >> 1;
  const int aCol = (tid & 1) * 16;
  // B: 32 k-rows x 64 halfs; thread owns k=tid>>3, n block [(tid&7)*8, +8).
  const int bK8 = tid >> 3;
  const int bN8 = (tid & 7) * 8;

  const _Float16* Ab   = A + (size_t)tileM * lda;        // advances 32/step
  const float*    B32b = Bf32 ? Bf32 + tileN : nullptr;  // advances 32*ldb
  const _Float16* B16b = Bf16 ? Bf16 + tileN : nullptr;

  auto stage = [&](int buf) {
    // A: 2x global b128 -> 2x LDS b128 (row-contiguous, no transpose)
    const _Float16* as = Ab + (size_t)aRow * lda + aCol;
    uint4 q0 = *(const uint4*)(as);
    uint4 q1 = *(const uint4*)(as + 8);
    _Float16* ad = AsAll[buf] + aRow * 32 + aCol;
    *(uint4*)(ad)     = q0;
    *(uint4*)(ad + 8) = q1;
    // B: wide loads, transpose-scatter into BsT[n][k]
    _Float16* bd = BsAll[buf];
    if (B32b) {
      const float* bs = B32b + (size_t)bK8 * ldb + bN8;
      float4 f0 = *(const float4*)(bs);
      float4 f1 = *(const float4*)(bs + 4);
      bd[(bN8 + 0) * 32 + bK8] = (_Float16)f0.x;
      bd[(bN8 + 1) * 32 + bK8] = (_Float16)f0.y;
      bd[(bN8 + 2) * 32 + bK8] = (_Float16)f0.z;
      bd[(bN8 + 3) * 32 + bK8] = (_Float16)f0.w;
      bd[(bN8 + 4) * 32 + bK8] = (_Float16)f1.x;
      bd[(bN8 + 5) * 32 + bK8] = (_Float16)f1.y;
      bd[(bN8 + 6) * 32 + bK8] = (_Float16)f1.z;
      bd[(bN8 + 7) * 32 + bK8] = (_Float16)f1.w;
    } else {
      Q16 p;
      p.q = *(const uint4*)(B16b + (size_t)bK8 * ldb + bN8);
      #pragma unroll
      for (int i = 0; i < 8; ++i) bd[(bN8 + i) * 32 + bK8] = p.h[i];
    }
  };

  v8f c00 = {}, c01 = {}, c10 = {}, c11 = {};

  const int nk = K >> 5;
  stage(0);
  Ab += 32;
  if (B32b) B32b += (size_t)32 * ldb; else B16b += (size_t)32 * ldb;
  __syncthreads();

  for (int kt = 0; kt < nk; ++kt) {
    const int cur = kt & 1;
    if (kt + 1 < nk) {
      // prefetch the tile after the one being staged (gfx1250 global_prefetch)
      if (kt + 2 < nk) {
        __builtin_prefetch(Ab + 32 + (size_t)aRow * lda + aCol, 0, 1);
        if (B32b) __builtin_prefetch(B32b + (size_t)(32 + bK8) * ldb + bN8, 0, 1);
        else      __builtin_prefetch(B16b + (size_t)(32 + bK8) * ldb + bN8, 0, 1);
      }
      stage(cur ^ 1);
      Ab += 32;
      if (B32b) B32b += (size_t)32 * ldb; else B16b += (size_t)32 * ldb;
    }

    // ---- fragments per the CDNA5 16-bit per-lane K layout ----
    Frag16 a0, a1, b0, b1;
    const uint32_t* ap0 = (const uint32_t*)(AsAll[cur] + (rm0 + lrow) * 32);
    const uint32_t* ap1 = (const uint32_t*)(AsAll[cur] + (rm0 + 16 + lrow) * 32);
    const uint32_t* bp0 = (const uint32_t*)(BsAll[cur] + (cn0 + lrow) * 32);
    const uint32_t* bp1 = (const uint32_t*)(BsAll[cur] + (cn0 + 16 + lrow) * 32);
    #pragma unroll
    for (int j = 0; j < 4; ++j) {
      a0.u[j] = ap0[kb2 + j]; a0.u[4 + j] = ap0[kb2 + 8 + j];
      a1.u[j] = ap1[kb2 + j]; a1.u[4 + j] = ap1[kb2 + 8 + j];
      b0.u[j] = bp0[kb2 + j]; b0.u[4 + j] = bp0[kb2 + 8 + j];
      b1.u[j] = bp1[kb2 + j]; b1.u[4 + j] = bp1[kb2 + 8 + j];
    }
    c00 = __builtin_amdgcn_wmma_f32_16x16x32_f16(false, a0.v, false, b0.v,
                                                 (short)0, c00, false, false);
    c01 = __builtin_amdgcn_wmma_f32_16x16x32_f16(false, a0.v, false, b1.v,
                                                 (short)0, c01, false, false);
    c10 = __builtin_amdgcn_wmma_f32_16x16x32_f16(false, a1.v, false, b0.v,
                                                 (short)0, c10, false, false);
    c11 = __builtin_amdgcn_wmma_f32_16x16x32_f16(false, a1.v, false, b1.v,
                                                 (short)0, c11, false, false);
    __syncthreads();
  }

  // ---- epilogue: lane (l<16 -> M=j, N=l) / (l>=16 -> M=8+j, N=l-16) ----
  const int nloc = lane & 15;
  const int mb   = (lane < 16) ? 0 : 8;
  #pragma unroll
  for (int ns = 0; ns < 2; ++ns) {
    const int col = tileN + cn0 + ns * 16 + nloc;
    const float bv = bias ? bias[col] : 0.f;
    #pragma unroll
    for (int ms = 0; ms < 2; ++ms) {
      v8f cc = ns ? (ms ? c11 : c01) : (ms ? c10 : c00);
      #pragma unroll
      for (int j = 0; j < 8; ++j) {
        int row = tileM + rm0 + ms * 16 + mb + j;
        float v = cc[j] + bv;
        if (act == 1)      v = fmaxf(v, 0.f);
        else if (act == 2) v = (v > 0.f) ? v : (__expf(v) - 1.f);
        size_t idx = (size_t)row * ldc + col;
        if (act == 3) v = fmaxf(v, 0.f) * (float)C[idx];
        C[idx] = (_Float16)v;
      }
    }
  }
}

// ---------------------------------------------------------------------------
// Points encoder: per node 2->64(BN,ReLU)->128(BN,ReLU)->256, L2-normalize.
// One 256-thread block per node. Writes FEATS[n][0:256]=descs, [256:512]=ep.
// ---------------------------------------------------------------------------
__global__ __launch_bounds__(256)
void k_pe(const float* __restrict__ pts, const float* __restrict__ descs,
          const float* __restrict__ W0, const float* __restrict__ W1,
          const float* __restrict__ W2,
          const float* __restrict__ b0, const float* __restrict__ b1,
          const float* __restrict__ b2,
          const float* __restrict__ g0, const float* __restrict__ bt0,
          const float* __restrict__ m0, const float* __restrict__ v0,
          const float* __restrict__ g1, const float* __restrict__ bt1,
          const float* __restrict__ m1, const float* __restrict__ v1,
          _Float16* __restrict__ feats) {
  __shared__ float h1[64];
  __shared__ float h2[128];
  __shared__ float h3[256];
  __shared__ float red[256];
  const int node = blockIdx.x;   // g*1024+n, 0..8191
  const int t = threadIdx.x;
  const float x0 = pts[(size_t)node * 2 + 0];
  const float x1 = pts[(size_t)node * 2 + 1];
  if (t < 64) {
    float a = x0 * W0[t] + x1 * W0[64 + t] + b0[t];
    a = g0[t] * (a - m0[t]) * rsqrtf(v0[t] + EPS_BN) + bt0[t];
    h1[t] = fmaxf(a, 0.f);
  }
  __syncthreads();
  if (t < 128) {
    float a = b1[t];
    for (int i = 0; i < 64; ++i) a += h1[i] * W1[i * 128 + t];
    a = g1[t] * (a - m1[t]) * rsqrtf(v1[t] + EPS_BN) + bt1[t];
    h2[t] = fmaxf(a, 0.f);
  }
  __syncthreads();
  {
    float a = b2[t];
    for (int i = 0; i < 128; ++i) a += h2[i] * W2[i * 256 + t];
    h3[t] = a;
    red[t] = a * a;
  }
  __syncthreads();
  for (int s = 128; s > 0; s >>= 1) {
    if (t < s) red[t] += red[t + s];
    __syncthreads();
  }
  const float inv = 1.f / fmaxf(sqrtf(red[0]), 1e-12f);
  _Float16* f = feats + (size_t)node * 512;
  f[t]       = (_Float16)descs[(size_t)node * 256 + t];
  f[256 + t] = (_Float16)(h3[t] * inv);
}

// f1 = Wh@a[:H], f2 = Wh@a[H:]  for nheads*1024 rows (one thread per row)
__global__ __launch_bounds__(256)
void k_f12(const _Float16* __restrict__ Wh, const float* __restrict__ a,
           float* __restrict__ f12, int H) {
  const int idx = blockIdx.x * 256 + threadIdx.x;  // h*1024 + n
  const int h = idx >> 10;
  const _Float16* w = Wh + (size_t)idx * H;
  const float* ah = a + (size_t)h * 2 * H;
  float s1 = 0.f, s2 = 0.f;
  for (int i = 0; i < H; ++i) {
    float wv = (float)w[i];
    s1 += wv * ah[i];
    s2 += wv * ah[H + i];
  }
  f12[(size_t)idx * 2 + 0] = s1;
  f12[(size_t)idx * 2 + 1] = s2;
}

// Masked LeakyReLU softmax row -> f16 attention.  grid (1024 rows, nheads)
__global__ __launch_bounds__(256)
void k_att(const float* __restrict__ adj, const float* __restrict__ f12,
           _Float16* __restrict__ att) {
  __shared__ float red[256];
  const int i = blockIdx.x;
  const int h = blockIdx.y;
  const int t = threadIdx.x;
  const float* arow = adj + (size_t)i * 1024;
  const float* fh   = f12 + (size_t)h * 1024 * 2;
  const float f1i = fh[(size_t)i * 2];
  float mx = -3.0e38f;
  for (int j = t; j < 1024; j += 256) {
    float e = f1i + fh[(size_t)j * 2 + 1];
    e = (e > 0.f) ? e : 0.2f * e;
    e = (arow[j] > 0.f) ? e : -9e15f;
    mx = fmaxf(mx, e);
  }
  red[t] = mx; __syncthreads();
  for (int s = 128; s > 0; s >>= 1) { if (t < s) red[t] = fmaxf(red[t], red[t + s]); __syncthreads(); }
  mx = red[0]; __syncthreads();
  float sum = 0.f;
  for (int j = t; j < 1024; j += 256) {
    float e = f1i + fh[(size_t)j * 2 + 1];
    e = (e > 0.f) ? e : 0.2f * e;
    e = (arow[j] > 0.f) ? e : -9e15f;
    sum += __expf(e - mx);
  }
  red[t] = sum; __syncthreads();
  for (int s = 128; s > 0; s >>= 1) { if (t < s) red[t] += red[t + s]; __syncthreads(); }
  const float inv = 1.f / red[0];
  _Float16* orow = att + ((size_t)h * 1024 + i) * 1024;
  for (int j = t; j < 1024; j += 256) {
    float e = f1i + fh[(size_t)j * 2 + 1];
    e = (e > 0.f) ? e : 0.2f * e;
    e = (arow[j] > 0.f) ? e : -9e15f;
    orow[j] = (_Float16)(__expf(e - mx) * inv);
  }
}

__global__ void k_zero(float* p, int n) {
  int i = blockIdx.x * 256 + threadIdx.x;
  if (i < n) p[i] = 0.f;
}

// Column sums of NODE[8192][1024] (f16) into S[1024] (f32, atomics).
__global__ __launch_bounds__(256)
void k_colsum(const _Float16* __restrict__ node, float* __restrict__ S) {
  const int t = threadIdx.x;
  const int b = blockIdx.x;  // 32 blocks x 256 rows
  float acc0 = 0.f, acc1 = 0.f, acc2 = 0.f, acc3 = 0.f;
  for (int r = b * 256; r < b * 256 + 256; ++r) {
    const _Float16* row = node + (size_t)r * 1024;
    acc0 += (float)row[t];
    acc1 += (float)row[t + 256];
    acc2 += (float)row[t + 512];
    acc3 += (float)row[t + 768];
  }
  atomicAdd(&S[t], acc0);
  atomicAdd(&S[t + 256], acc1);
  atomicAdd(&S[t + 512], acc2);
  atomicAdd(&S[t + 768], acc3);
}

// Shifted column sums S_k from total S and boundary rows (mean∘conv collapse).
__global__ void k_sk(const float* __restrict__ S,
                     const _Float16* __restrict__ node,
                     float* __restrict__ Sk) {
  const int c = blockIdx.x * 256 + threadIdx.x;  // 0..1023
  const float e0 = (float)node[c];
  const float e1 = (float)node[1024 + c];
  const float e2 = (float)node[(size_t)8190 * 1024 + c];
  const float e3 = (float)node[(size_t)8191 * 1024 + c];
  const float s = S[c];
  Sk[c]        = s - e2 - e3;  // k=0: rows 0..8189
  Sk[1024 + c] = s - e0 - e3;  // k=1: rows 1..8190
  Sk[2048 + c] = s - e0 - e1;  // k=2: rows 2..8191
}

// y[o] = b[o] + (1/8190) * sum_{c,k} K[o,c,k] * S_k[c]
__global__ __launch_bounds__(256)
void k_tcn(const float* __restrict__ Kw, const float* __restrict__ tb,
           const float* __restrict__ Sk, float* __restrict__ Y) {
  const int o = blockIdx.x * 256 + threadIdx.x;  // 0..2047
  const float* ko = Kw + (size_t)o * 3072;
  float acc = 0.f;
  for (int c = 0; c < 1024; ++c) {
    acc += ko[c * 3 + 0] * Sk[c]
         + ko[c * 3 + 1] * Sk[1024 + c]
         + ko[c * 3 + 2] * Sk[2048 + c];
  }
  Y[o] = acc * (1.f / 8190.f) + tb[o];
}

__global__ __launch_bounds__(256)
void k_final(const float* __restrict__ Y, float* __restrict__ out) {
  __shared__ float red[256];
  const int t = threadIdx.x;
  float s = 0.f;
  for (int o = t; o < 2048; o += 256) { float y = Y[o]; s += y * y; }
  red[t] = s; __syncthreads();
  for (int st = 128; st > 0; st >>= 1) { if (t < st) red[t] += red[t + st]; __syncthreads(); }
  const float inv = 1.f / fmaxf(sqrtf(red[0]), 1e-12f);
  for (int o = t; o < 2048; o += 256) out[o] = Y[o] * inv;
}

// ---------------------------------------------------------------------------
static inline void gemm(hipStream_t s, const _Float16* A, const float* B32,
                        const _Float16* B16, const float* bias, _Float16* C,
                        int M, int N, int K, int lda, int ldb, int ldc,
                        long long sA, long long sB, long long sC, int Z, int act) {
  dim3 g(N / 64, M / 128, Z);
  k_gemm<<<g, 256, 0, s>>>(A, B32, B16, bias, C, M, N, K, lda, ldb, ldc,
                           sA, sB, sC, act);
}

extern "C" void kernel_launch(void* const* d_in, const int* in_sizes, int n_in,
                              void* d_out, int out_size, void* d_ws, size_t ws_size,
                              hipStream_t stream) {
  // ---- input leaves in setup_inputs() insertion order ----
  const float* P_pts   = (const float*)d_in[0];   // [8,1024,2]
  const float* P_descs = (const float*)d_in[1];   // [8,1024,256]
  const float* P_adj   = (const float*)d_in[2];   // [8,1024,1024]
  const float* peW0 = (const float*)d_in[3];
  const float* peW1 = (const float*)d_in[4];
  const float* peW2 = (const float*)d_in[5];
  const float* peb0 = (const float*)d_in[6];
  const float* peb1 = (const float*)d_in[7];
  const float* peb2 = (const float*)d_in[8];
  const float* bn0g = (const float*)d_in[9];
  const float* bn0b = (const float*)d_in[10];
  const float* bn0m = (const float*)d_in[11];
  const float* bn0v = (const float*)d_in[12];
  const float* bn1g = (const float*)d_in[13];
  const float* bn1b = (const float*)d_in[14];
  const float* bn1m = (const float*)d_in[15];
  const float* bn1v = (const float*)d_in[16];
  const float* gatW[2]  = { (const float*)d_in[17], (const float*)d_in[21] };
  const float* gatA[2]  = { (const float*)d_in[18], (const float*)d_in[22] };
  const float* gatWo[2] = { (const float*)d_in[19], (const float*)d_in[23] };
  const float* gatAo[2] = { (const float*)d_in[20], (const float*)d_in[24] };
  const float* tran1_W = (const float*)d_in[25];
  const float* tran1_b = (const float*)d_in[26];
  const float* fe1_W = (const float*)d_in[27];
  const float* fe1_b = (const float*)d_in[28];
  const float* fe2_W = (const float*)d_in[29];
  const float* fe2_b = (const float*)d_in[30];
  const float* le1_W = (const float*)d_in[31];
  const float* le1_b = (const float*)d_in[32];
  const float* le2_W = (const float*)d_in[33];
  const float* le2_b = (const float*)d_in[34];
  const float* tcn_K = (const float*)d_in[35];
  const float* tcn_b = (const float*)d_in[36];

  // ---- scratch layout ----
  char* w = (char*)d_ws;
  size_t off = 0;
  auto take = [&](size_t bytes) -> void* {
    size_t o = (off + 255) & ~(size_t)255;
    off = o + bytes;
    return (void*)(w + o);
  };
  _Float16* FEATS   = (_Float16*)take(8ull * 1024 * 512 * 2);   // feats / X3
  _Float16* XG1     = (_Float16*)take(8ull * 1024 * 512 * 2);   // gat1 out / D1
  _Float16* XG2     = (_Float16*)take(8ull * 1024 * 512 * 2);   // gat2 out / L1
  _Float16* WH      = (_Float16*)take(8ull * 1024 * 64 * 2);    // per-head Wh, reused as Wh_o
  _Float16* HEADCAT = (_Float16*)take(1024ull * 512 * 2);
  _Float16* ATT     = (_Float16*)take(8ull * 1024 * 1024 * 2);
  float*    F12     = (float*)take(8ull * 1024 * 2 * 4);
  _Float16* NODE    = (_Float16*)take(8ull * 1024 * 1024 * 2);
  float*    S       = (float*)take(1024 * 4);
  float*    SK      = (float*)take(3 * 1024 * 4);
  float*    Y       = (float*)take(2048 * 4);
  _Float16* X3 = FEATS;
  _Float16* D1 = XG1;
  _Float16* L1 = XG2;

  // ---- points encoder: fill FEATS (descs | normalized ep) as f16 ----
  k_pe<<<8192, 256, 0, stream>>>(P_pts, P_descs, peW0, peW1, peW2,
                                 peb0, peb1, peb2,
                                 bn0g, bn0b, bn0m, bn0v,
                                 bn1g, bn1b, bn1m, bn1v, FEATS);

  // ---- two stacked GAT layers per graph ----
  for (int g = 0; g < 8; ++g) {
    const float* adjg = P_adj + (size_t)g * 1024 * 1024;
    for (int layer = 0; layer < 2; ++layer) {
      const _Float16* X = (layer == 0 ? FEATS : XG1) + (size_t)g * 1024 * 512;
      _Float16* Out     = (layer == 0 ? XG1 : XG2) + (size_t)g * 1024 * 512;
      // 8 heads batched: Wh_h = X @ W_h   [1024,512]x[512,64]
      gemm(stream, X, gatW[layer], nullptr, nullptr, WH,
           1024, 64, 512, 512, 64, 64, 0, 512 * 64, 1024 * 64, 8, 0);
      k_f12<<<32, 256, 0, stream>>>(WH, gatA[layer], F12, 64);
      k_att<<<dim3(1024, 8), 256, 0, stream>>>(adjg, F12, ATT);
      // head outputs (ELU) into HEADCAT column blocks: att_h @ Wh_h
      gemm(stream, ATT, nullptr, WH, nullptr, HEADCAT,
           1024, 64, 1024, 1024, 64, 512,
           1024ll * 1024, 1024ll * 64, 64, 8, 2);
      // output attention layer: Wh_o = HEADCAT @ Wo  [1024,512]x[512,512]
      gemm(stream, HEADCAT, gatWo[layer], nullptr, nullptr, WH,
           1024, 512, 512, 512, 512, 512, 0, 0, 0, 1, 0);
      k_f12<<<4, 256, 0, stream>>>(WH, gatAo[layer], F12, 512);
      k_att<<<dim3(1024, 1), 256, 0, stream>>>(adjg, F12, ATT);
      // Out = ELU(att @ Wh_o)
      gemm(stream, ATT, nullptr, WH, nullptr, Out,
           1024, 512, 1024, 1024, 512, 512, 0, 0, 0, 1, 2);
    }
  }

  // ---- post-GAT linears, batched across all 8 graphs ----
  gemm(stream, XG2, tran1_W, nullptr, tran1_b, X3,
       1024, 512, 512, 512, 512, 512, 1024ll * 512, 0, 1024ll * 512, 8, 1);
  gemm(stream, X3, fe1_W, nullptr, fe1_b, D1,
       1024, 512, 512, 512, 512, 512, 1024ll * 512, 0, 1024ll * 512, 8, 1);
  gemm(stream, D1, fe2_W, nullptr, fe2_b, NODE,
       1024, 1024, 512, 512, 1024, 1024, 1024ll * 512, 0, 1024ll * 1024, 8, 1);
  gemm(stream, X3, le1_W, nullptr, le1_b, L1,
       1024, 512, 512, 512, 512, 512, 1024ll * 512, 0, 1024ll * 512, 8, 1);
  // NODE = relu(L1@le2+b) * NODE   (fused sparsification product)
  gemm(stream, L1, le2_W, nullptr, le2_b, NODE,
       1024, 1024, 512, 512, 1024, 1024, 1024ll * 512, 0, 1024ll * 1024, 8, 3);

  // ---- collapsed TCN: mean∘conv == weighted shifted column sums ----
  k_zero<<<4, 256, 0, stream>>>(S, 1024);
  k_colsum<<<32, 256, 0, stream>>>(NODE, S);
  k_sk<<<4, 256, 0, stream>>>(S, NODE, SK);
  k_tcn<<<8, 256, 0, stream>>>(tcn_K, tcn_b, SK, Y);
  k_final<<<1, 256, 0, stream>>>(Y, (float*)d_out);
}
